// GatedGraphNeuralNetwork_74491912781909
// MI455X (gfx1250) — matrix operand order repaired and checked by reading, not compiled, not run on hardware
//
#include <hip/hip_runtime.h>

// ---------------------------------------------------------------------------
// Gated Graph NN on MI455X (gfx1250, wave32, WMMA, async-to-LDS staging)
//   N=100000 nodes, E=600000 edges, H=128, 5 GRU steps.
// h kept in fp32 (GRU blend) + bf16 mirror (WMMA A staging via
// global_load_async_to_lds_b128). agg fp32 (atomic scatter target, L2-resident).
// ---------------------------------------------------------------------------

#define NN 100000
#define NE 600000
#define HD 128
#define SA_STRIDE 68   // dwords per staged LDS row (64 data + 4 pad -> no bank conflicts)

typedef __attribute__((ext_vector_type(16))) __bf16        v16bf;
typedef __attribute__((ext_vector_type(8)))  float         v8f;
typedef __attribute__((ext_vector_type(4)))  unsigned int  u32x4;
typedef __attribute__((ext_vector_type(4)))  float         f32x4;

union FragU { u32x4 u[2]; v16bf v; };

static __device__ __forceinline__ unsigned int f2bf(float f) {
  unsigned int u = __float_as_uint(f);
  return (u + 0x7FFFu + ((u >> 16) & 1u)) >> 16;  // RNE fp32->bf16
}
static __device__ __forceinline__ unsigned int packbf(float a, float b) {
  return f2bf(a) | (f2bf(b) << 16);
}

static __device__ __forceinline__ v8f wmma_bf16(v16bf a, v16bf b, v8f c) {
  return __builtin_amdgcn_wmma_f32_16x16x32_bf16(false, a, false, b, (short)0, c,
                                                 false, false);
}

// Async DMA one 16B chunk per lane: LDS[ldsa] = MEM[sbase + voff]  (GVS mode).
static __device__ __forceinline__ void async_b128(unsigned ldsa, unsigned voff,
                                                  unsigned long long sbase) {
  asm volatile("global_load_async_to_lds_b128 %0, %1, %2"
               :: "v"(ldsa), "v"(voff), "s"(sbase)
               : "memory");
}
static __device__ __forceinline__ void wait_async0() {
  asm volatile("s_wait_asynccnt 0" ::: "memory");
}

// B-fragment (32x16 bf16 K-chunk), prepacked:
// dword idx = (((kc*NT + nt)*2 + khalf)*16 + n)*8 + j  -> 8 contiguous dwords/lane.
static __device__ __forceinline__ v16bf load_bfrag(const unsigned int* __restrict__ q,
                                                   int NT, int kc, int nt, int lane) {
  const int n  = lane & 15;
  const int kh = lane >> 4;
  const int base = (((kc * NT + nt) * 2 + kh) * 16 + n) * 8;
  FragU f;
  f.u[0] = *(const u32x4*)(q + base);
  f.u[1] = *(const u32x4*)(q + base + 4);
  return f.v;
}

// A-fragment (16x32 bf16) from LDS rows, contiguous k = 16*khalf + s (same order as B packing).
static __device__ __forceinline__ void load_afrag(FragU& f, const unsigned int* sRow,
                                                  int kc, int khalf) {
  f.u[0] = *(const u32x4*)(sRow + kc * 16 + khalf * 8);
  f.u[1] = *(const u32x4*)(sRow + kc * 16 + khalf * 8 + 4);
}

static __device__ __forceinline__ float sigmoidf_fast(float x) {
  return 1.0f / (1.0f + __expf(-x));
}
static __device__ __forceinline__ float tanhf_fast(float x) {
  return 2.0f / (1.0f + __expf(-2.0f * x)) - 1.0f;
}

// ---------------------------------------------------------------------------
// Kernel 0a: repack W^T into bf16 fragment layout. B[k][col] = W[col][k].
// ---------------------------------------------------------------------------
__global__ __launch_bounds__(256) void ggnn_prep_weights(
    const float* __restrict__ Wmsg, const float* __restrict__ Wih,
    const float* __restrict__ Whh, unsigned int* __restrict__ Qmsg,
    unsigned int* __restrict__ Qih, unsigned int* __restrict__ Qhh) {
  int tid = blockIdx.x * blockDim.x + threadIdx.x;
  const float* W;
  unsigned int* Q;
  int NT, d;
  if (tid < 8192)                { W = Wmsg; Q = Qmsg; NT = 8;  d = tid; }
  else if (tid < 8192 + 24576)   { W = Wih;  Q = Qih;  NT = 24; d = tid - 8192; }
  else if (tid < 8192 + 49152)   { W = Whh;  Q = Qhh;  NT = 24; d = tid - 8192 - 24576; }
  else return;
  int j  = d & 7;  int r = d >> 3;
  int n  = r & 15; r >>= 4;
  int kh = r & 1;  r >>= 1;
  int nt = r % NT; int kc = r / NT;
  int k   = kc * 32 + kh * 16 + 2 * j;
  int col = nt * 16 + n;
  Q[d] = packbf(W[col * HD + k], W[col * HD + k + 1]);
}

// Kernel 0b: build bf16 mirror of h (from x) -- one packed dword per thread.
__global__ __launch_bounds__(256) void ggnn_pack_h(const float* __restrict__ src,
                                                   unsigned int* __restrict__ dst) {
  int i = blockIdx.x * blockDim.x + threadIdx.x;
  if (i < NN * HD / 2) dst[i] = packbf(src[2 * i], src[2 * i + 1]);
}

// ---------------------------------------------------------------------------
// Kernel 1: fused gather -> (h[col] @ W_msg^T + b_msg) -> atomic scatter-add.
// Gather staged via async-to-LDS DMA from the bf16 h mirror.
// ---------------------------------------------------------------------------
__global__ __launch_bounds__(256) void ggnn_msg_scatter(
    const unsigned short* __restrict__ hbf,   // bf16 mirror [NN][HD]
    const long long* __restrict__ eidx,       // int64 [2][NE]
    const unsigned int* __restrict__ Qmsg, const float* __restrict__ bmsg,
    float* __restrict__ agg) {
  __shared__ __align__(16) unsigned int sA[128 * SA_STRIDE];

  const int t = threadIdx.x;
  const long long eBase = (long long)blockIdx.x * 128;
  const bool full = (eBase + 128) <= (long long)NE;  // block-uniform

  // ---- stage 128 gathered bf16 rows into LDS via async DMA (no VALU) ----
  {
    const int eloc = t >> 1;
    const int half = t & 1;
    long long e = eBase + eloc;
    if (e >= NE) e = NE - 1;                 // clamp (still a valid gather)
    const int src = (int)eidx[NE + e];       // edge_index[1] = source
    const unsigned gvoff = (unsigned)src * (HD * 2) + (unsigned)half * 128;
    const unsigned ldsa =
        (unsigned)(size_t)(const void*)&sA[eloc * SA_STRIDE + half * 32];
    const unsigned long long base = (unsigned long long)hbf;
#pragma unroll
    for (int j = 0; j < 8; ++j)
      async_b128(ldsa + j * 16, gvoff + j * 16, base);
  }
  wait_async0();
  __syncthreads();

  const int lane  = t & 31;
  const int wv    = t >> 5;
  const int mrow  = lane & 15;
  const int khalf = lane >> 4;
  const int ncol  = lane & 15;
  const unsigned int* sRow = &sA[(wv * 16 + mrow) * SA_STRIDE];

  FragU A[4];
#pragma unroll
  for (int kc = 0; kc < 4; ++kc) load_afrag(A[kc], sRow, kc, khalf);

  // ---- 8 x (16x16) output tiles, biased ----
  v8f acc[8];
#pragma unroll
  for (int nt = 0; nt < 8; ++nt) {
    v8f c = {};
#pragma unroll
    for (int kc = 0; kc < 4; ++kc)
      c = wmma_bf16(A[kc].v, load_bfrag(Qmsg, 8, kc, nt, lane), c);
    const float bias = bmsg[nt * 16 + ncol];
#pragma unroll
    for (int i = 0; i < 8; ++i) c[i] += bias;
    acc[nt] = c;
  }

  // ---- destination rows: clamped load (branch-free), zero tail lanes ----
  int dn[8];
#pragma unroll
  for (int i = 0; i < 8; ++i) {
    long long e = eBase + wv * 16 + khalf * 8 + i;
    if (e >= NE) e = NE - 1;
    dn[i] = (int)eidx[e];                    // edge_index[0] = destination
  }
  if (!full) {                               // rare: 1 of 4688 blocks
#pragma unroll
    for (int i = 0; i < 8; ++i)
      if (eBase + wv * 16 + khalf * 8 + i >= NE)
#pragma unroll
        for (int nt = 0; nt < 8; ++nt) acc[nt][i] = 0.0f;  // +0.0 is a no-op add
  }

  // ---- single shared scatter loop (64 f32 atomics/lane, all L2) ----
#pragma unroll
  for (int nt = 0; nt < 8; ++nt)
#pragma unroll
    for (int i = 0; i < 8; ++i)
      atomicAdd(&agg[dn[i] * HD + nt * 16 + ncol], acc[nt][i]);
}

// ---------------------------------------------------------------------------
// Kernel 2: fused GRU cell. sH staged via async DMA from bf16 mirror;
// sAgg converted fp32->bf16 (atomic target must stay fp32).
// Writes fp32 h_new and refreshes the bf16 mirror (own rows only -> race-free).
// ---------------------------------------------------------------------------
__global__ __launch_bounds__(256) void ggnn_gru_update(
    const float* __restrict__ hprev, const float* __restrict__ agg,
    const unsigned int* __restrict__ Qih, const unsigned int* __restrict__ Qhh,
    const float* __restrict__ bih, const float* __restrict__ bhh,
    float* __restrict__ hout, unsigned short* __restrict__ hbf) {
  __shared__ __align__(16) unsigned int sAgg[128 * SA_STRIDE];
  __shared__ __align__(16) unsigned int sH[128 * SA_STRIDE];

  const int t = threadIdx.x;
  const long long nodeBase = (long long)blockIdx.x * 128;
  const bool fullN = (nodeBase + 128) <= (long long)NN;  // block-uniform

  {
    const int nloc = t >> 1;
    const int half = t & 1;
    long long nd = nodeBase + nloc;
    if (nd >= NN) nd = NN - 1;
    // h rows: async DMA from bf16 mirror
    {
      const unsigned gvoff = (unsigned)nd * (HD * 2) + (unsigned)half * 128;
      const unsigned ldsa =
          (unsigned)(size_t)(const void*)&sH[nloc * SA_STRIDE + half * 32];
      const unsigned long long base = (unsigned long long)hbf;
#pragma unroll
      for (int j = 0; j < 8; ++j)
        async_b128(ldsa + j * 16, gvoff + j * 16, base);
    }
    // agg rows: fp32 -> bf16 conversion
    const f32x4* arow = (const f32x4*)(agg + nd * HD + half * 64);
    unsigned int* da = &sAgg[nloc * SA_STRIDE + half * 32];
#pragma unroll
    for (int jj = 0; jj < 8; ++jj) {
      f32x4 a0 = arow[2 * jj], a1 = arow[2 * jj + 1];
      u32x4 pa;
      pa.x = packbf(a0.x, a0.y); pa.y = packbf(a0.z, a0.w);
      pa.z = packbf(a1.x, a1.y); pa.w = packbf(a1.z, a1.w);
      *(u32x4*)(da + jj * 4) = pa;
    }
  }
  wait_async0();
  __syncthreads();

  const int lane  = t & 31;
  const int wv    = t >> 5;
  const int mrow  = lane & 15;
  const int khalf = lane >> 4;
  const int ncol  = lane & 15;
  const unsigned int* sRa = &sAgg[(wv * 16 + mrow) * SA_STRIDE];
  const unsigned int* sRh = &sH[(wv * 16 + mrow) * SA_STRIDE];

  FragU Aa[4], Ah[4];
#pragma unroll
  for (int kc = 0; kc < 4; ++kc) {
    load_afrag(Aa[kc], sRa, kc, khalf);
    load_afrag(Ah[kc], sRh, kc, khalf);
  }

  // clamped per-lane row offsets (int: < 2^31), hoisted out of the tile loop
  int rowoff[8];
#pragma unroll
  for (int i = 0; i < 8; ++i) {
    long long node = nodeBase + wv * 16 + khalf * 8 + i;
    if (node >= NN) node = NN - 1;
    rowoff[i] = (int)node * HD;
  }

#pragma unroll
  for (int nt = 0; nt < 8; ++nt) {
    v8f ar = {}, az = {}, ain = {}, ahn = {};
#pragma unroll
    for (int kc = 0; kc < 4; ++kc) {
      ar  = wmma_bf16(Aa[kc].v, load_bfrag(Qih, 24, kc, nt,      lane), ar);
      ar  = wmma_bf16(Ah[kc].v, load_bfrag(Qhh, 24, kc, nt,      lane), ar);
      az  = wmma_bf16(Aa[kc].v, load_bfrag(Qih, 24, kc, nt + 8,  lane), az);
      az  = wmma_bf16(Ah[kc].v, load_bfrag(Qhh, 24, kc, nt + 8,  lane), az);
      ain = wmma_bf16(Aa[kc].v, load_bfrag(Qih, 24, kc, nt + 16, lane), ain);
      ahn = wmma_bf16(Ah[kc].v, load_bfrag(Qhh, 24, kc, nt + 16, lane), ahn);
    }
    const int c = nt * 16 + ncol;
    const float br  = bih[c] + bhh[c];
    const float bz  = bih[HD + c] + bhh[HD + c];
    const float bni = bih[2 * HD + c];
    const float bnh = bhh[2 * HD + c];

    // gate math once (clamped loads), only the stores take the uniform branch
    float hnew[8];
#pragma unroll
    for (int i = 0; i < 8; ++i) {
      const float hold = hprev[rowoff[i] + c];
      const float r  = sigmoidf_fast(ar[i] + br);
      const float z  = sigmoidf_fast(az[i] + bz);
      const float nv = tanhf_fast(ain[i] + bni + r * (ahn[i] + bnh));
      hnew[i] = (1.0f - z) * nv + z * hold;
    }
    if (fullN) {
#pragma unroll
      for (int i = 0; i < 8; ++i) {
        hout[rowoff[i] + c] = hnew[i];
        hbf[rowoff[i] + c] = (unsigned short)f2bf(hnew[i]);
      }
    } else {                                 // rare: 1 of 782 blocks
#pragma unroll
      for (int i = 0; i < 8; ++i)
        if (nodeBase + wv * 16 + khalf * 8 + i < NN) {
          hout[rowoff[i] + c] = hnew[i];
          hbf[rowoff[i] + c] = (unsigned short)f2bf(hnew[i]);
        }
    }
  }
}

// ---------------------------------------------------------------------------
// Host launch.
// ---------------------------------------------------------------------------
extern "C" void kernel_launch(void* const* d_in, const int* in_sizes, int n_in,
                              void* d_out, int out_size, void* d_ws, size_t ws_size,
                              hipStream_t stream) {
  (void)in_sizes; (void)n_in; (void)out_size; (void)ws_size;

  const float* x        = (const float*)d_in[0];
  const long long* eidx = (const long long*)d_in[1];  // int64 [2, E]
  const float* Wmsg     = (const float*)d_in[2];
  const float* bmsg     = (const float*)d_in[3];
  const float* Wih      = (const float*)d_in[4];
  const float* Whh      = (const float*)d_in[5];
  const float* bih      = (const float*)d_in[6];
  const float* bhh      = (const float*)d_in[7];
  float* hout = (float*)d_out;

  char* w = (char*)d_ws;
  const size_t nhBytes  = (size_t)NN * HD * sizeof(float);  // 51.2 MB
  const size_t bfBytes  = (size_t)NN * HD * 2;              // 25.6 MB
  float*          agg  = (float*)w;
  float*          hbuf = (float*)(w + nhBytes);
  unsigned short* hbf  = (unsigned short*)(w + 2 * nhBytes);
  unsigned int*   qmsg = (unsigned int*)(w + 2 * nhBytes + bfBytes);
  unsigned int*   qih  = (unsigned int*)(w + 2 * nhBytes + bfBytes + 8192u * 4u);
  unsigned int*   qhh  = (unsigned int*)(w + 2 * nhBytes + bfBytes + 8192u * 4u + 24576u * 4u);

  ggnn_prep_weights<<<224, 256, 0, stream>>>(Wmsg, Wih, Whh, qmsg, qih, qhh);
  ggnn_pack_h<<<(NN * HD / 2 + 255) / 256, 256, 0, stream>>>(x, (unsigned int*)hbf);

  const int msgGrid = (NE + 127) / 128;  // 4688
  const int gruGrid = (NN + 127) / 128;  // 782

  const float* hsrc = x;
  for (int step = 0; step < 5; ++step) {
    float* hdst = (step & 1) ? hbuf : hout;  // steps 0,2,4 -> d_out (final there)
    hipMemsetAsync(agg, 0, nhBytes, stream);
    ggnn_msg_scatter<<<msgGrid, 256, 0, stream>>>(hbf, eidx, qmsg, bmsg, agg);
    ggnn_gru_update<<<gruGrid, 256, 0, stream>>>(hsrc, agg, qih, qhh, bih, bhh,
                                                 hdst, hbf);
    hsrc = hdst;
  }
}
